// CustomCausalAttention_17832704213564
// MI455X (gfx1250) — matrix operand order repaired and checked by reading, not compiled, not run on hardware
//
#include <hip/hip_runtime.h>

#define NB 2
#define NH 16
#define N_SEQ 2048
#define D_DIM 128
#define BQ 128          // query rows per block (8 waves x 16)
#define BK 32           // key rows per chunk
#define SM_SCALE 0.08838834764831845f

typedef __attribute__((ext_vector_type(16))) __bf16 v16bf;
typedef __attribute__((ext_vector_type(8)))  float  v8f;
typedef __attribute__((ext_vector_type(4)))  unsigned int v4u;
typedef __attribute__((ext_vector_type(8)))  int    v8i;
typedef __attribute__((ext_vector_type(4)))  int    v4i;

union ABFrag { v16bf v; uint4 u[2]; };

// ---------------------------------------------------------------------------
// LDS byte-offset of a __shared__ object (generic -> addrspace(3) -> offset)
// ---------------------------------------------------------------------------
__device__ __forceinline__ uint32_t lds_off_u32(const void* p) {
    return (uint32_t)(size_t)(__attribute__((address_space(3))) const void*)p;
}

// ---------------------------------------------------------------------------
// Tensor Data Mover: DMA a 2D bf16 tile (tile_d1 rows x tile_d0 elems) from a
// row-major [tensor_d1 x tensor_d0] global tensor (row stride stride0 elems)
// into contiguous LDS at lds_off. D# bit layout per CDNA5 ISA 8.3/8.4.
// ---------------------------------------------------------------------------
__device__ __forceinline__ void tdm_load_2d(uint32_t lds_off, const void* gptr,
                                            uint32_t tensor_d0, uint32_t tensor_d1,
                                            uint32_t tile_d0,   uint32_t tile_d1,
                                            uint32_t stride0)
{
    const unsigned long long ga = (unsigned long long)(size_t)gptr;
    v4u g0;
    g0[0] = 1u;                                              // count=1, user D#
    g0[1] = lds_off;                                         // lds_addr [63:32]
    g0[2] = (uint32_t)ga;                                    // global_addr lo
    g0[3] = (uint32_t)((ga >> 32) & 0x1FFFFFFu) | (2u << 30);// addr hi | type=2
    v8i g1;
    g1[0] = (int)0x00010000u;                                // data_size=1 (2B), mask=0
    g1[1] = (int)((tensor_d0 & 0xFFFFu) << 16);              // tensor_dim0 [79:48]
    g1[2] = (int)((tensor_d0 >> 16) | ((tensor_d1 & 0xFFFFu) << 16)); // dim0 hi | dim1 lo
    g1[3] = (int)((tensor_d1 >> 16) | (tile_d0 << 16));      // dim1 hi | tile_dim0
    g1[4] = (int)(tile_d1 & 0xFFFFu);                        // tile_dim1 (tile_dim2=0)
    g1[5] = (int)stride0;                                    // tensor_dim0_stride lo
    g1[6] = 0;                                               // stride0 hi | stride1 lo
    g1[7] = 0;                                               // stride1 hi
    const v4i z4 = {0, 0, 0, 0};                             // groups 2/3: 2D tensor
    const v8i z8 = {0, 0, 0, 0, 0, 0, 0, 0};                 // unused extra group slot
    __builtin_amdgcn_tensor_load_to_lds(g0, g1, z4, z4, z8, 0);
}

// ---------------------------------------------------------------------------
// Pre-pass 1: fp32 -> bf16 for Q,K plus row norms s*||q||^2, s*||k||^2.
// One block (128 threads) per row of length D=128.
// ---------------------------------------------------------------------------
__global__ __launch_bounds__(128) void rbf_prep_kernel(
    const float* __restrict__ q, const float* __restrict__ k,
    __bf16* __restrict__ qb, __bf16* __restrict__ kb,
    float* __restrict__ qsq, float* __restrict__ ksq)
{
    const int row = blockIdx.x;               // 0 .. B*H*N-1
    const int t   = threadIdx.x;              // 0..127
    const size_t idx = (size_t)row * D_DIM + t;

    float qf = q[idx], kf = k[idx];
    qb[idx] = (__bf16)qf;
    kb[idx] = (__bf16)kf;

    float qq = qf * qf, kk = kf * kf;
    #pragma unroll
    for (int off = 16; off > 0; off >>= 1) {
        qq += __shfl_down(qq, off, 32);
        kk += __shfl_down(kk, off, 32);
    }
    __shared__ float rq[4], rk[4];
    const int w = t >> 5, lane = t & 31;
    if (lane == 0) { rq[w] = qq; rk[w] = kk; }
    __syncthreads();
    if (t == 0) {
        qsq[row] = SM_SCALE * (rq[0] + rq[1] + rq[2] + rq[3]);
        ksq[row] = SM_SCALE * (rk[0] + rk[1] + rk[2] + rk[3]);
    }
}

// ---------------------------------------------------------------------------
// Pre-pass 2: V fp32 [bh][n][d] -> bf16 transposed vt[bh][d][n] (32x32 tiles)
// grid = (B*H, N/32, D/32), block = 256
// ---------------------------------------------------------------------------
__global__ __launch_bounds__(256) void rbf_vt_kernel(
    const float* __restrict__ v, __bf16* __restrict__ vt)
{
    __shared__ __bf16 tl[32][33];
    const int bh = blockIdx.x, nt = blockIdx.y, dt = blockIdx.z;
    const int t  = threadIdx.x;
    {
        const int r  = t >> 3;              // n within tile
        const int c4 = (t & 7) * 4;         // d within tile
        const size_t src = ((size_t)bh * N_SEQ + nt * 32 + r) * D_DIM + dt * 32 + c4;
        float4 f = *(const float4*)(v + src);
        tl[c4 + 0][r] = (__bf16)f.x;
        tl[c4 + 1][r] = (__bf16)f.y;
        tl[c4 + 2][r] = (__bf16)f.z;
        tl[c4 + 3][r] = (__bf16)f.w;
    }
    __syncthreads();
    {
        const int dr = t >> 3;              // d within tile
        const int n4 = (t & 7) * 4;         // n within tile
        const size_t dst = ((size_t)bh * D_DIM + dt * 32 + dr) * N_SEQ + nt * 32 + n4;
        vt[dst + 0] = tl[dr][n4 + 0];
        vt[dst + 1] = tl[dr][n4 + 1];
        vt[dst + 2] = tl[dr][n4 + 2];
        vt[dst + 3] = tl[dr][n4 + 3];
    }
}

// ---------------------------------------------------------------------------
// Main kernel: flash-style causal RBF attention, bf16 WMMA + TDM double buffer.
// grid = (B*H, N/BQ), block = 256 (8 wave32); wave w owns 16 query rows.
// ---------------------------------------------------------------------------
__global__ __launch_bounds__(256) void rbf_attn_kernel(
    const __bf16* __restrict__ qb, const __bf16* __restrict__ kb,
    const __bf16* __restrict__ vt, const float* __restrict__ qsq,
    const float* __restrict__ ksq, float* __restrict__ out)
{
    __shared__ __attribute__((aligned(16))) __bf16 sK [2][BK * D_DIM];  // [n][d]
    __shared__ __attribute__((aligned(16))) __bf16 sVt[2][D_DIM * BK];  // [d][n]
    __shared__ __attribute__((aligned(16))) __bf16 sP [8][16 * BK];     // per-wave P

    const int bh   = blockIdx.x;
    const int qblk = blockIdx.y;
    const int tid  = threadIdx.x;
    const int w    = tid >> 5;
    const int l    = tid & 31;
    const int lh   = l >> 4;                  // half-wave 0/1
    const int ll   = l & 15;
    const int c0   = lh * 8;                  // A-operand K-interleave base

    const int    qm0     = qblk * BQ + w * 16;
    const size_t baseRow = (size_t)bh * N_SEQ;
    const __bf16* vtb    = vt + (size_t)bh * D_DIM * N_SEQ;

    // ---- Q fragments in A-operand layout: 4 chunks of K=32 over D=128 ----
    ABFrag qa[4];
    {
        const __bf16* qrow = qb + (baseRow + qm0 + ll) * D_DIM;
        #pragma unroll
        for (int kc = 0; kc < 4; ++kc) {
            const int d0 = kc * 32 + c0;
            qa[kc].u[0] = *(const uint4*)(qrow + d0);
            qa[kc].u[1] = *(const uint4*)(qrow + d0 + 16);
        }
    }
    float qsqm[8];
    #pragma unroll
    for (int vv = 0; vv < 8; ++vv)
        qsqm[vv] = qsq[baseRow + qm0 + vv + 8 * lh];

    v8f oacc[8];
    #pragma unroll
    for (int dc = 0; dc < 8; ++dc) oacc[dc] = {};

    const int nChunks = (qblk + 1) * (BQ / BK);

    // ---- prologue: TDM-stage chunk 0 into buffer 0 (wave 0 issues) ----
    if (w == 0) {
        tdm_load_2d(lds_off_u32(&sK[0][0]),  kb + baseRow * D_DIM,
                    D_DIM, N_SEQ, D_DIM, BK, D_DIM);
        tdm_load_2d(lds_off_u32(&sVt[0][0]), vtb,
                    N_SEQ, D_DIM, BK, D_DIM, N_SEQ);
    }

    for (int ck = 0; ck < nChunks; ++ck) {
        const int kv0 = ck * BK;
        const int cur = ck & 1;

        if (w == 0) __builtin_amdgcn_s_wait_tensorcnt((short)0);
        __syncthreads();                       // buffer `cur` ready, buffer `1-cur` free

        // ---- issue next chunk's TDM into the other buffer (overlaps compute) ----
        if (w == 0 && (ck + 1) < nChunks) {
            const int kvn = kv0 + BK;
            tdm_load_2d(lds_off_u32(&sK[1 - cur][0]),
                        kb + (baseRow + kvn) * D_DIM,
                        D_DIM, N_SEQ, D_DIM, BK, D_DIM);
            tdm_load_2d(lds_off_u32(&sVt[1 - cur][0]),
                        vtb + kvn,
                        N_SEQ, D_DIM, BK, D_DIM, N_SEQ);
        }

        if (kv0 >= qm0 + 16) continue;         // strictly-future chunk (wave-uniform)

        // ---- S = Q K^T : two 16x16 key tiles, K-dim 128 in 4 wmmas each ----
        v8f s0 = {}, s1 = {};
        #pragma unroll
        for (int kc = 0; kc < 4; ++kc) {
            ABFrag b0, b1;
            const int d0 = kc * 32 + lh * 8;   // B-operand: K-interleaved like A (8h, 16+8h)
            b0.u[0] = *(const uint4*)(sK[cur] + ll * D_DIM + d0);
            b0.u[1] = *(const uint4*)(sK[cur] + ll * D_DIM + d0 + 16);
            b1.u[0] = *(const uint4*)(sK[cur] + (ll + 16) * D_DIM + d0);
            b1.u[1] = *(const uint4*)(sK[cur] + (ll + 16) * D_DIM + d0 + 16);
            s0 = __builtin_amdgcn_wmma_f32_16x16x32_bf16(false, qa[kc].v, false, b0.v,
                                                         (short)0, s0, false, false);
            s1 = __builtin_amdgcn_wmma_f32_16x16x32_bf16(false, qa[kc].v, false, b1.v,
                                                         (short)0, s1, false, false);
        }

        // ---- p = exp(2s*qk - s||q||^2 - s||k||^2), causal mask, C->A transpose ----
        const float ksq0 = ksq[baseRow + kv0 + ll];
        const float ksq1 = ksq[baseRow + kv0 + ll + 16];
        #pragma unroll
        for (int vv = 0; vv < 8; ++vv) {
            const int m  = vv + 8 * lh;
            const int mg = qm0 + m;
            {
                const int   ng = kv0 + ll;
                const float lg = 2.0f * SM_SCALE * s0[vv] - qsqm[vv] - ksq0;
                const float p  = (ng <= mg) ? __expf(lg) : 0.0f;
                sP[w][m * BK + ll] = (__bf16)p;
            }
            {
                const int   ng = kv0 + ll + 16;
                const float lg = 2.0f * SM_SCALE * s1[vv] - qsqm[vv] - ksq1;
                const float p  = (ng <= mg) ? __expf(lg) : 0.0f;
                sP[w][m * BK + ll + 16] = (__bf16)p;
            }
        }
        __builtin_amdgcn_wave_barrier();       // same-wave DS ops are in-order
        asm volatile("" ::: "memory");

        ABFrag pa;
        pa.u[0] = *(const uint4*)(sP[w] + ll * BK + c0);
        pa.u[1] = *(const uint4*)(sP[w] + ll * BK + c0 + 16);

        // ---- O += P * V over 8 d-tiles of 16 ----
        #pragma unroll
        for (int dc = 0; dc < 8; ++dc) {
            ABFrag vf;
            const __bf16* col = sVt[cur] + (dc * 16 + ll) * BK + lh * 8;
            vf.u[0] = *(const uint4*)(col);
            vf.u[1] = *(const uint4*)(col + 16);
            oacc[dc] = __builtin_amdgcn_wmma_f32_16x16x32_bf16(false, pa.v, false, vf.v,
                                                               (short)0, oacc[dc], false, false);
        }
    }

    // ---- epilogue: fp32 output, C/D layout -> [bh][row][d] ----
    #pragma unroll
    for (int dc = 0; dc < 8; ++dc) {
        #pragma unroll
        for (int vv = 0; vv < 8; ++vv) {
            const int m = vv + 8 * lh;
            out[(baseRow + qm0 + m) * D_DIM + dc * 16 + ll] = oacc[dc][vv];
        }
    }
}

// ---------------------------------------------------------------------------
extern "C" void kernel_launch(void* const* d_in, const int* in_sizes, int n_in,
                              void* d_out, int out_size, void* d_ws, size_t ws_size,
                              hipStream_t stream) {
    const float* q = (const float*)d_in[0];
    const float* k = (const float*)d_in[1];
    const float* v = (const float*)d_in[2];
    float* out = (float*)d_out;

    const size_t nElem = (size_t)NB * NH * N_SEQ * D_DIM;   // 8,388,608
    const size_t nRows = (size_t)NB * NH * N_SEQ;           // 65,536

    __bf16* qb  = (__bf16*)d_ws;
    __bf16* kb  = qb + nElem;
    __bf16* vt  = kb + nElem;
    float*  qsq = (float*)(vt + nElem);
    float*  ksq = qsq + nRows;

    rbf_prep_kernel<<<dim3((unsigned)nRows), dim3(128), 0, stream>>>(
        q, k, qb, kb, qsq, ksq);
    rbf_vt_kernel<<<dim3(NB * NH, N_SEQ / 32, D_DIM / 32), dim3(256), 0, stream>>>(
        v, vt);

    dim3 grid(NB * NH, N_SEQ / BQ);
    rbf_attn_kernel<<<grid, dim3(256), 0, stream>>>(qb, kb, vt, qsq, ksq, out);
}
